// DEMainModalityMILMoE_79551384257141
// MI455X (gfx1250) — matrix operand order repaired and checked by reading, not compiled, not run on hardware
//
#include <hip/hip_runtime.h>

// ---------------------------------------------------------------------------
// DEMainModalityMILMoE — exact algebraic reduction for MI455X (gfx1250).
//
// Exact identities (hold for ALL inputs):
//  * _dense_cross with a length-1 kv sequence: softmax over 1 key == 1.0 and
//    there is no residual, so the result is ((kv@wv+bv)@wo+bo)@wp+bp,
//    independent of the query. cross_rna overwrites cross_text, so the
//    backbone input depends ONLY on rna + cross_rna params. => img projection
//    (68.7 GFLOP), scorer, top-k, gather, text path, cross_text: dead code.
//  * With all 256 tokens identical, deform attention's sampled values all
//    equal x and the softmax weights sum to 1 => attn out == concat_H(x);
//    realized as an A-matrix wrapping mod 256 in K. Tokens stay identical
//    through every layer; mean-pool is the identity.
//
// Remaining work: 16 batch-vectors of 256 through the layer chain. All GEMMs
// have M=16 — one WMMA tile row — and run on v_wmma_f32_16x16x32_bf16 with
// fp32 accumulate, converting fp32 A/W to bf16 in registers (each weight is
// used exactly once per launch, so a staged bf16 copy would only add
// traffic). ~22 MB weights / ~0.17 GFLOP: ~1 us of HBM time at 23.3 TB/s.
//
// N / LDA / ACT are template constants so the B-stream lowers to clauses of
// immediate-offset global_load_b32 with one pointer bump per K-step.
// ---------------------------------------------------------------------------

typedef __bf16 v16bf __attribute__((ext_vector_type(16)));
typedef float  v8f   __attribute__((ext_vector_type(8)));

#define ACT_NONE 0
#define ACT_RELU 1
#define ACT_GELU 2

// C[16,N] = act(A[16,K] @ W[K,N] + bias)
//  - A row-major, physical row stride LDA (power of two). If LDA < K the A
//    columns wrap modulo LDA (folds the H-tiled deform projection).
//  - W row-major [K,N] fp32 -> bf16 in registers.
//  - One wave (32 threads) per 16-column tile; K multiple of 32.
template <int N, int LDA, int ACT>
__global__ void gemm16_wmma_bf16(const float* __restrict__ A,
                                 const float* __restrict__ W,
                                 const float* __restrict__ bias,
                                 float* __restrict__ out,
                                 int K) {
  constexpr int KMASK = LDA - 1;
  const int lane  = threadIdx.x;      // 0..31
  const int kh    = lane >> 4;        // wave half
  const int ln15  = lane & 15;
  const int ntile = blockIdx.x * 16;
  const int nCol  = ntile + ln15;
  const bool nValid = (N % 16 == 0) || (nCol < N);

  const float* Arow = A + ln15 * LDA;
  // Column base; invalid lanes clamp to column 0 (their D column is never
  // stored and cannot contaminate valid columns).
  const float* Wcol = W + (nValid ? nCol : 0) + (size_t)kh * 16 * N;

  v8f acc = {};
  for (int kb = 0; kb < K; kb += 32) {
    // A fragment (16x32 bf16): half 0 -> K = kb+0..15,
    // half 1 -> K = kb+16..31; row m = lane&15.
    v16bf afrag;
    {
      const int ka0 = (kb + kh * 16) & KMASK;       // 8-aligned, runs stay
      const int ka1 = (kb + kh * 16 + 8) & KMASK;   // contiguous under mask
#pragma unroll
      for (int j = 0; j < 8; ++j) afrag[j]     = (__bf16)Arow[ka0 + j];
#pragma unroll
      for (int j = 0; j < 8; ++j) afrag[8 + j] = (__bf16)Arow[ka1 + j];
    }
    // B fragment (32x16 bf16): half 0 -> K = kb+0..15, half 1 -> kb+16..31;
    // column n = ntile + (lane&15). j*N are compile-time immediates.
    v16bf bfrag;
#pragma unroll
    for (int j = 0; j < 16; ++j) bfrag[j] = (__bf16)Wcol[(size_t)j * N];
    if (kb + 64 < K)  // prefetch 2 K-steps ahead of the weight stream
      __builtin_prefetch(Wcol + (size_t)64 * N, 0, 1);
    Wcol += (size_t)32 * N;

    acc = __builtin_amdgcn_wmma_f32_16x16x32_bf16(
        /*neg_a=*/false, afrag, /*neg_b=*/false, bfrag,
        /*c_mod=*/(short)0, acc, /*reuse_a=*/false, /*reuse_b=*/false);
  }

  // D layout: VGPR r -> row m = r + half*8, column n = ntile + (lane&15).
  if (nValid) {
    const float bv = bias ? bias[nCol] : 0.0f;
#pragma unroll
    for (int r = 0; r < 8; ++r) {
      const int m = r + kh * 8;
      float v = acc[r] + bv;
      if (ACT == ACT_RELU)      v = fmaxf(v, 0.0f);
      else if (ACT == ACT_GELU) v = 0.5f * v * (1.0f + erff(v * 0.70710678118654752f));
      out[(size_t)m * N + nCol] = v;
    }
  }
}

// out = LayerNorm(x + a) * g + b over last dim (256). One block (8 waves) per
// row; wave32 shuffle reduction + one LDS stage across waves.
__global__ void ln_residual_kernel(const float* __restrict__ x,
                                   const float* __restrict__ a,
                                   const float* __restrict__ g,
                                   const float* __restrict__ b,
                                   float* __restrict__ out) {
  __shared__ float partial[8];
  const int t = threadIdx.x;
  const int base = blockIdx.x * 256;
  const float v = x[base + t] + a[base + t];

  float s = v;
#pragma unroll
  for (int m = 16; m > 0; m >>= 1) s += __shfl_xor(s, m, 32);
  if ((t & 31) == 0) partial[t >> 5] = s;
  __syncthreads();
  float tot = 0.0f;
#pragma unroll
  for (int w = 0; w < 8; ++w) tot += partial[w];
  const float mean = tot * (1.0f / 256.0f);
  const float d = v - mean;

  __syncthreads();  // all partial[] reads done before reuse
  float q = d * d;
#pragma unroll
  for (int m = 16; m > 0; m >>= 1) q += __shfl_xor(q, m, 32);
  if ((t & 31) == 0) partial[t >> 5] = q;
  __syncthreads();
  float vtot = 0.0f;
#pragma unroll
  for (int w = 0; w < 8; ++w) vtot += partial[w];
  const float var = vtot * (1.0f / 256.0f);

  out[base + t] = d * rsqrtf(var + 1e-5f) * g[t] + b[t];
}

// hazards = sigmoid(logits); survival = cumprod(1 - hazards) along PRED=20.
__global__ void finalize_kernel(const float* __restrict__ logits,
                                float* __restrict__ outp) {
  const int bidx = threadIdx.x;
  if (bidx < 16) {
    float s = 1.0f;
    for (int j = 0; j < 20; ++j) {
      const float h = 1.0f / (1.0f + expf(-logits[bidx * 20 + j]));
      outp[bidx * 20 + j] = h;              // hazards [16,20]
      s *= (1.0f - h);
      outp[320 + bidx * 20 + j] = s;        // survival [16,20]
    }
  }
}

extern "C" void kernel_launch(void* const* d_in, const int* in_sizes, int n_in,
                              void* d_out, int out_size, void* d_ws, size_t ws_size,
                              hipStream_t stream) {
  (void)in_sizes; (void)n_in; (void)out_size; (void)ws_size;

  // setup_inputs() insertion-order flattening:
  // 0 img, 1 text, 2 rna,
  // 3 img_proj.w, 4 img_proj.b, 5 text_proj.w, 6 text_proj.b,
  // 7 rna_proj.w, 8 rna_proj.b, 9 scorer.w, 10 scorer.b,
  // 11..20 cross_text {wq,bq,wk,bk,wv,bv,wo,bo,wp,bp},
  // 21..30 cross_rna  {wq,bq,wk,bk,wv,bv,wo,bo,wp,bp},
  // 31 off_w, 32 off_b, 33 att_w, 34 att_b, 35 proj_w, 36 proj_b,
  // 37 ln1_g, 38 ln1_b, 39 w1, 40 b1, 41 w2, 42 b2, 43 ln2_g, 44 ln2_b,
  // 45 head.w, 46 head.b
  const float* rna    = (const float*)d_in[2];   // [16,2048]
  const float* rna_w  = (const float*)d_in[7];   // [2048,256]
  const float* rna_b  = (const float*)d_in[8];
  const float* wv     = (const float*)d_in[25];  // cross_rna V / O / P chain
  const float* bv     = (const float*)d_in[26];
  const float* wo     = (const float*)d_in[27];
  const float* bo     = (const float*)d_in[28];
  const float* wp     = (const float*)d_in[29];
  const float* bp     = (const float*)d_in[30];
  const float* proj_w = (const float*)d_in[35];  // [6,1024,256]
  const float* proj_b = (const float*)d_in[36];  // [6,256]
  const float* ln1_g  = (const float*)d_in[37];
  const float* ln1_b  = (const float*)d_in[38];
  const float* w1     = (const float*)d_in[39];  // [6,256,1024]
  const float* b1     = (const float*)d_in[40];  // [6,1024]
  const float* w2     = (const float*)d_in[41];  // [6,1024,256]
  const float* b2     = (const float*)d_in[42];  // [6,256]
  const float* ln2_g  = (const float*)d_in[43];
  const float* ln2_b  = (const float*)d_in[44];
  const float* head_w = (const float*)d_in[45];  // [256,20]
  const float* head_b = (const float*)d_in[46];  // [20]

  float* wsf = (float*)d_ws;
  float* X0 = wsf;            // [16,256]
  float* X1 = X0 + 16 * 256;  // [16,256]
  float* T0 = X1 + 16 * 256;  // [16,1024] FFN hidden
  float* T2 = T0 + 16 * 1024; // [16,256]  attn-out / FFN-out temp
  float* LG = T2 + 16 * 256;  // [16,20]   logits

  const dim3 wave(32);

  // x0 = relu(rna @ rna_proj.w + b)
  gemm16_wmma_bf16<256, 2048, ACT_RELU><<<dim3(16), wave, 0, stream>>>(
      rna, rna_w, rna_b, X0, 2048);
  // cross_rna (query-independent): x = ((x0@wv+bv)@wo+bo)@wp+bp
  gemm16_wmma_bf16<256, 256, ACT_NONE><<<dim3(16), wave, 0, stream>>>(
      X0, wv, bv, X1, 256);
  gemm16_wmma_bf16<256, 256, ACT_NONE><<<dim3(16), wave, 0, stream>>>(
      X1, wo, bo, X0, 256);
  gemm16_wmma_bf16<256, 256, ACT_NONE><<<dim3(16), wave, 0, stream>>>(
      X0, wp, bp, X1, 256);

  float* cur = X1;
  float* alt = X0;
  for (int i = 0; i < 6; ++i) {
    // deform attn collapsed: a = concat_H(x) @ proj_w[i] + proj_b[i]
    // (A wraps mod 256 in K via LDA=256 < K=1024)
    gemm16_wmma_bf16<256, 256, ACT_NONE><<<dim3(16), wave, 0, stream>>>(
        cur, proj_w + (size_t)i * 1024 * 256, proj_b + i * 256, T2, 1024);
    ln_residual_kernel<<<dim3(16), dim3(256), 0, stream>>>(
        cur, T2, ln1_g + i * 256, ln1_b + i * 256, alt);
    { float* t = cur; cur = alt; alt = t; }
    // FFN: h = gelu(x@w1+b1); f = h@w2+b2; x = LN(x + f)
    gemm16_wmma_bf16<1024, 256, ACT_GELU><<<dim3(64), wave, 0, stream>>>(
        cur, w1 + (size_t)i * 256 * 1024, b1 + i * 1024, T0, 256);
    gemm16_wmma_bf16<256, 1024, ACT_NONE><<<dim3(16), wave, 0, stream>>>(
        T0, w2 + (size_t)i * 1024 * 256, b2 + i * 256, T2, 1024);
    ln_residual_kernel<<<dim3(16), dim3(256), 0, stream>>>(
        cur, T2, ln2_g + i * 256, ln2_b + i * 256, alt);
    { float* t = cur; cur = alt; alt = t; }
  }

  // head (N=20, masked tail tile) + sigmoid/cumprod epilogue
  gemm16_wmma_bf16<20, 256, ACT_NONE><<<dim3(2), wave, 0, stream>>>(
      cur, head_w, head_b, LG, 256);
  finalize_kernel<<<dim3(1), dim3(32), 0, stream>>>(LG, (float*)d_out);
}
